// DSTformer_35682588295527
// MI455X (gfx1250) — compile-verified, matrix operand order checked by reading, not run on hardware
//
#include <hip/hip_runtime.h>
#include <hip/hip_bf16.h>

typedef __attribute__((ext_vector_type(16))) _Float16 v16h;
typedef __attribute__((ext_vector_type(8)))  _Float16 v8h;
typedef __attribute__((ext_vector_type(8)))  float    v8f;

#define B_     4
#define LSEQ   2048
#define DMODEL 512
#define DINNER 1024
#define DSTATE 16
#define DTRANK 32
#define DCONV  4
#define MROWS  (B_ * LSEQ)          // 8192

// ---------------------------------------------------------------------------
// fp32 -> fp16 conversion (weights + activations feeding WMMA)
// ---------------------------------------------------------------------------
__global__ void k_cvt_f16(const float* __restrict__ in, _Float16* __restrict__ out, int n) {
    int i = blockIdx.x * blockDim.x + threadIdx.x;
    if (i < n) out[i] = (_Float16)in[i];
}

// ---------------------------------------------------------------------------
// Register-blocked WMMA GEMM: C(M,N) = A(M,K) * W(N,K)^T
// [+ optional softplus(bias) epilogue for dt_proj]
//
// N, K are compile-time so that (a) the K-loop has a static trip count (no
// zero-trip guard / acc re-zero branch), and (b) epilogue row strides fold
// into the 24-bit instruction offset of global_store_b32 (one address VGPR
// per 16x16 tile instead of per-store 64-bit adds).
//
// Each wave computes an (MT*16 x NT*16) output block: MT A-fragments +
// NT B-fragments feed MT*NT WMMAs per k-step. Block = 256 threads = 8 waves
// in a 2(M) x 4(N) wave grid.
//
// Fragment layout per CDNA5 ISA 7.12.2 (16-bit A 16x32):
//   lane 0..15  : row M=lane,    halves = K {0..7} then {16..23}
//   lane 16..31 : row M=lane-16, halves = K {8..15} then {24..31}
// (B operand mirrors A; W stored row-major over N, K.)
// C layout: VGPR v -> row = v + 8*(lane>=16), col = lane&15.
// ---------------------------------------------------------------------------
template <int MT, int NT, int EPI, int N, int K>
__global__ __launch_bounds__(256)
void k_wmma_gemm(const _Float16* __restrict__ A, const _Float16* __restrict__ W,
                 const float* __restrict__ bias, float* __restrict__ C) {
    const int tid  = threadIdx.x;
    const int lane = tid & 31;
    const int l15  = lane & 15;
    const int hi   = lane >> 4;
    const int wave = tid >> 5;          // 0..7
    const int wm   = wave >> 2;         // 0..1
    const int wn   = wave & 3;          // 0..3
    const int tileM = (blockIdx.y * 2 + wm) * (16 * MT);
    const int tileN = (blockIdx.x * 4 + wn) * (16 * NT);

    const _Float16* ap[MT];
    const _Float16* bp[NT];
#pragma unroll
    for (int mi = 0; mi < MT; ++mi)
        ap[mi] = A + (tileM + 16 * mi + l15) * K + 8 * hi;
#pragma unroll
    for (int ni = 0; ni < NT; ++ni)
        bp[ni] = W + (tileN + 16 * ni + l15) * K + 8 * hi;

    v8f acc[MT][NT] = {};

#pragma unroll 2
    for (int kk = 0; kk < K; kk += 32) {
        // prefetch next k-slice through the gfx1250 prefetch path
#pragma unroll
        for (int mi = 0; mi < MT; ++mi) __builtin_prefetch(ap[mi] + kk + 32, 0, 1);
#pragma unroll
        for (int ni = 0; ni < NT; ++ni) __builtin_prefetch(bp[ni] + kk + 32, 0, 1);

        v16h av[MT], bv[NT];
#pragma unroll
        for (int mi = 0; mi < MT; ++mi) {
            v8h lo  = *(const v8h*)(ap[mi] + kk);
            v8h hi8 = *(const v8h*)(ap[mi] + kk + 16);
#pragma unroll
            for (int i = 0; i < 8; ++i) { av[mi][i] = lo[i]; av[mi][i + 8] = hi8[i]; }
        }
#pragma unroll
        for (int ni = 0; ni < NT; ++ni) {
            v8h lo  = *(const v8h*)(bp[ni] + kk);
            v8h hi8 = *(const v8h*)(bp[ni] + kk + 16);
#pragma unroll
            for (int i = 0; i < 8; ++i) { bv[ni][i] = lo[i]; bv[ni][i + 8] = hi8[i]; }
        }
#pragma unroll
        for (int mi = 0; mi < MT; ++mi)
#pragma unroll
            for (int ni = 0; ni < NT; ++ni)
                acc[mi][ni] = __builtin_amdgcn_wmma_f32_16x16x32_f16(
                                  false, av[mi], false, bv[ni],
                                  (short)0, acc[mi][ni], false, false);
    }

#pragma unroll
    for (int mi = 0; mi < MT; ++mi) {
#pragma unroll
        for (int ni = 0; ni < NT; ++ni) {
            const int col = tileN + 16 * ni + l15;
            float bcol = (EPI == 1) ? bias[col] : 0.f;
            float* cptr = C + (tileM + 16 * mi + 8 * hi) * N + col;
#pragma unroll
            for (int v = 0; v < 8; ++v) {
                float val = acc[mi][ni][v];
                if (EPI == 1) {                      // softplus(acc + b_dt)
                    val += bcol;
                    val = (val > 20.f) ? val : log1pf(__expf(val));
                }
                cptr[v * N] = val;                   // constant-folded offsets
            }
        }
    }
}

// ---------------------------------------------------------------------------
// Depthwise causal conv1d (d_conv=4) + bias + SiLU.
// x comes from the first DINNER columns of xz; emits u in f32 (scan) and f16 (x_proj A).
// ---------------------------------------------------------------------------
__global__ void k_conv_silu(const float* __restrict__ xz, const float* __restrict__ cw,
                            const float* __restrict__ cb, float* __restrict__ u32,
                            _Float16* __restrict__ u16) {
    int idx = blockIdx.x * blockDim.x + threadIdx.x;     // over MROWS*DINNER
    if (idx >= MROWS * DINNER) return;
    int d   = idx & (DINNER - 1);
    int row = idx >> 10;                                 // b*LSEQ + t
    int t   = row & (LSEQ - 1);
    float acc = cb[d];
#pragma unroll
    for (int j = 0; j < DCONV; ++j) {
        int tt = t - (DCONV - 1) + j;
        if (tt >= 0) {
            float xv = xz[(size_t)(row - (DCONV - 1) + j) * (2 * DINNER) + d];
            acc += cw[d * DCONV + j] * xv;
        }
    }
    float s = acc / (1.f + __expf(-acc));                // SiLU
    u32[idx] = s;
    u16[idx] = (_Float16)s;
}

// Extract dt_low (first 32 cols of x_dbl) to f16 for the dt_proj WMMA A operand.
__global__ void k_extract_dt16(const float* __restrict__ xdbl, _Float16* __restrict__ dt16, int n) {
    int idx = blockIdx.x * blockDim.x + threadIdx.x;     // over MROWS*DTRANK
    if (idx >= n) return;
    int r = idx >> 5, c = idx & 31;
    dt16[idx] = (_Float16)xdbl[(size_t)r * (DTRANK + 2 * DSTATE) + c];
}

// ---------------------------------------------------------------------------
// Selective scan: h_t = exp(dt*A)*h_{t-1} + (dt*u)*B_t ; y_t = <h_t, C_t>.
// One thread per (batch, channel); 16-state h in registers; B_t/C_t (shared by
// all channels of a batch) staged through LDS in 64-step chunks.
// ---------------------------------------------------------------------------
#define TCHUNK 64
__global__ __launch_bounds__(256)
void k_scan(const float* __restrict__ xdbl, const float* __restrict__ delta,
            const float* __restrict__ u32, const float* __restrict__ A_log,
            float* __restrict__ y) {
    __shared__ float sB[TCHUNK][DSTATE];
    __shared__ float sC[TCHUNK][DSTATE];
    const int b = blockIdx.x >> 2;
    const int d = ((blockIdx.x & 3) << 8) + threadIdx.x;

    float Ar[DSTATE], h[DSTATE];
#pragma unroll
    for (int n = 0; n < DSTATE; ++n) {
        Ar[n] = -__expf(A_log[d * DSTATE + n]);
        h[n]  = 0.f;
    }

    for (int t0 = 0; t0 < LSEQ; t0 += TCHUNK) {
        __syncthreads();
        for (int e = threadIdx.x; e < TCHUNK * DSTATE; e += 256) {
            int tt = e >> 4, n = e & 15;
            size_t base = ((size_t)b * LSEQ + t0 + tt) * (DTRANK + 2 * DSTATE);
            sB[tt][n] = xdbl[base + DTRANK + n];
            sC[tt][n] = xdbl[base + DTRANK + DSTATE + n];
        }
        __syncthreads();
        for (int t = 0; t < TCHUNK; ++t) {
            size_t ridx = ((size_t)b * LSEQ + t0 + t) * DINNER + d;
            float dt  = delta[ridx];
            float ut  = u32[ridx];
            float dtu = dt * ut;
            float yv  = 0.f;
#pragma unroll
            for (int n = 0; n < DSTATE; ++n) {
                h[n] = h[n] * __expf(dt * Ar[n]) + dtu * sB[t][n];
                yv  += h[n] * sC[t][n];
            }
            y[ridx] = yv;
        }
    }
}

// ---------------------------------------------------------------------------
// y' = (y + u*D) * silu(z)  -> f16 A operand for out_proj GEMM
// ---------------------------------------------------------------------------
__global__ void k_gate(const float* __restrict__ y, const float* __restrict__ u32,
                       const float* __restrict__ xz, const float* __restrict__ Dv,
                       _Float16* __restrict__ yg16) {
    int idx = blockIdx.x * blockDim.x + threadIdx.x;     // over MROWS*DINNER
    if (idx >= MROWS * DINNER) return;
    int d   = idx & (DINNER - 1);
    int row = idx >> 10;
    float zv = xz[(size_t)row * (2 * DINNER) + DINNER + d];
    float g  = zv / (1.f + __expf(-zv));
    float v  = (y[idx] + u32[idx] * Dv[d]) * g;
    yg16[idx] = (_Float16)v;
}

// ---------------------------------------------------------------------------
extern "C" void kernel_launch(void* const* d_in, const int* in_sizes, int n_in,
                              void* d_out, int out_size, void* d_ws, size_t ws_size,
                              hipStream_t stream) {
    const float* hidden = (const float*)d_in[0];   // (4,2048,512)
    const float* W_in   = (const float*)d_in[1];   // (2048,512)
    const float* conv_w = (const float*)d_in[2];   // (1024,1,4)
    const float* conv_b = (const float*)d_in[3];   // (1024,)
    const float* W_x    = (const float*)d_in[4];   // (64,1024)
    const float* W_dt   = (const float*)d_in[5];   // (1024,32)
    const float* b_dt   = (const float*)d_in[6];   // (1024,)
    const float* A_log  = (const float*)d_in[7];   // (1024,16)
    const float* Dv     = (const float*)d_in[8];   // (1024,)
    const float* W_out  = (const float*)d_in[9];   // (512,1024)
    float* out          = (float*)d_out;           // (4,2048,512)

    // workspace bump allocator (256B aligned)
    char*  base = (char*)d_ws;
    size_t off  = 0;
    auto alloc = [&](size_t bytes) -> void* {
        void* p = base + off;
        off += (bytes + 255) & ~(size_t)255;
        return p;
    };
    _Float16* hidden16 = (_Float16*)alloc((size_t)MROWS * DMODEL * 2);
    _Float16* Win16    = (_Float16*)alloc((size_t)(2 * DINNER) * DMODEL * 2);
    _Float16* Wx16     = (_Float16*)alloc((size_t)(DTRANK + 2 * DSTATE) * DINNER * 2);
    _Float16* Wdt16    = (_Float16*)alloc((size_t)DINNER * DTRANK * 2);
    _Float16* Wout16   = (_Float16*)alloc((size_t)DMODEL * DINNER * 2);
    float*    xz       = (float*)alloc((size_t)MROWS * 2 * DINNER * 4);
    float*    u32      = (float*)alloc((size_t)MROWS * DINNER * 4);
    _Float16* u16      = (_Float16*)alloc((size_t)MROWS * DINNER * 2);
    float*    xdbl     = (float*)alloc((size_t)MROWS * (DTRANK + 2 * DSTATE) * 4);
    _Float16* dt16     = (_Float16*)alloc((size_t)MROWS * DTRANK * 2);
    float*    delta    = (float*)alloc((size_t)MROWS * DINNER * 4);
    float*    yscan    = (float*)alloc((size_t)MROWS * DINNER * 4);
    _Float16* yg16     = (_Float16*)alloc((size_t)MROWS * DINNER * 2);

    const int TPB = 256;
    auto blks = [](size_t n, int t) { return (unsigned)((n + t - 1) / t); };

    // 1) f16 conversions (weights once + activations)
    k_cvt_f16<<<blks((size_t)MROWS * DMODEL, TPB), TPB, 0, stream>>>(hidden, hidden16, MROWS * DMODEL);
    k_cvt_f16<<<blks((size_t)2 * DINNER * DMODEL, TPB), TPB, 0, stream>>>(W_in, Win16, 2 * DINNER * DMODEL);
    k_cvt_f16<<<blks((size_t)(DTRANK + 2 * DSTATE) * DINNER, TPB), TPB, 0, stream>>>(W_x, Wx16, (DTRANK + 2 * DSTATE) * DINNER);
    k_cvt_f16<<<blks((size_t)DINNER * DTRANK, TPB), TPB, 0, stream>>>(W_dt, Wdt16, DINNER * DTRANK);
    k_cvt_f16<<<blks((size_t)DMODEL * DINNER, TPB), TPB, 0, stream>>>(W_out, Wout16, DMODEL * DINNER);

    // 2) in_proj: xz = hidden @ W_in^T   (8192 x 2048, K=512), 64x128 block tile
    {
        dim3 g((2 * DINNER) / 128, MROWS / 64);
        k_wmma_gemm<2, 2, 0, 2 * DINNER, DMODEL><<<g, 256, 0, stream>>>(hidden16, Win16, nullptr, xz);
    }

    // 3) depthwise causal conv + SiLU -> u
    k_conv_silu<<<blks((size_t)MROWS * DINNER, TPB), TPB, 0, stream>>>(xz, conv_w, conv_b, u32, u16);

    // 4) x_proj: x_dbl = u @ W_x^T   (8192 x 64, K=1024), 64x64 block tile
    {
        dim3 g((DTRANK + 2 * DSTATE) / 64, MROWS / 64);
        k_wmma_gemm<2, 1, 0, DTRANK + 2 * DSTATE, DINNER><<<g, 256, 0, stream>>>(u16, Wx16, nullptr, xdbl);
    }

    // 5) dt_proj + softplus fused epilogue: delta (8192 x 1024, K=32)
    k_extract_dt16<<<blks((size_t)MROWS * DTRANK, TPB), TPB, 0, stream>>>(xdbl, dt16, MROWS * DTRANK);
    {
        dim3 g(DINNER / 128, MROWS / 64);
        k_wmma_gemm<2, 2, 1, DINNER, DTRANK><<<g, 256, 0, stream>>>(dt16, Wdt16, b_dt, delta);
    }

    // 6) selective scan
    k_scan<<<B_ * (DINNER / 256), 256, 0, stream>>>(xdbl, delta, u32, A_log, yscan);

    // 7) D-skip + gate
    k_gate<<<blks((size_t)MROWS * DINNER, TPB), TPB, 0, stream>>>(yscan, u32, xz, Dv, yg16);

    // 8) out_proj: out = y' @ W_out^T   (8192 x 512, K=1024)
    {
        dim3 g(DMODEL / 128, MROWS / 64);
        k_wmma_gemm<2, 2, 0, DMODEL, DINNER><<<g, 256, 0, stream>>>(yg16, Wout16, nullptr, out);
    }
}
